// ResDetHeaBlk_v2_88983132438784
// MI455X (gfx1250) — matrix operand order repaired and checked by reading, not compiled, not run on hardware
//
#include <hip/hip_runtime.h>
#include <hip/hip_bf16.h>

// ---------------------------------------------------------------------------
// Types for CDNA5 WMMA (wave32): bf16 A/B fragments (v16bf), f32 acc (v8f)
// ---------------------------------------------------------------------------
typedef __attribute__((ext_vector_type(16))) __bf16 bf16x16;
typedef __attribute__((ext_vector_type(8)))  float  f32x8;

static __device__ __forceinline__ int imini(int a, int b) { return a < b ? a : b; }

// ---------------------------------------------------------------------------
// Implicit-GEMM convolution, bf16 WMMA, f32 accumulate.
//   Xp : activations, bf16 NHWC  [B, H, W, Cin]
//   Wf : weights in WMMA A-fragment layout:
//          [Cout/16][nK][lane(32)][16 halfs],  nK = KH*KW*Cin/32
//        lane = hl*16 + (co&15); element e holds K_local = hl*16 + e
//   Y  : output, f32 NCHW [B, Cout, H, W]
// Epilogue: y = acc * (gamma ? gamma[c]*rsqrt(1+1e-5) : 1) + (beta ? beta[c] : 0),
//           optional ReLU.
// Block: 128 threads = 4 waves. Each wave: 2 N-tiles (32 pixels) x 4 M-tiles
// (64 Cout rows) -> 8 WMMAs per K-step, one b256 load per fragment.
// Requires: Cin % 32 == 0, Cout % 64 == 0, (H*W) % 128 == 0.
// ---------------------------------------------------------------------------
__global__ __launch_bounds__(128)
void conv_wmma_kernel(const __bf16* __restrict__ Xp,
                      const __bf16* __restrict__ Wf,
                      float* __restrict__ Y,
                      const float* __restrict__ gamma,
                      const float* __restrict__ beta,
                      int relu, int Cin, int Cout, int H, int W, int KH, int KW)
{
    const int lane = threadIdx.x & 31;
    const int wav  = threadIdx.x >> 5;
    const int l15  = lane & 15;
    const int hl   = lane >> 4;
    const int b    = blockIdx.z;
    const int HW   = H * W;
    const int n0   = blockIdx.x * 128 + wav * 32;
    const int m0   = blockIdx.y * 64;
    const int padh = KH >> 1;
    const int padw = KW >> 1;
    const int KHW  = KH * KW;
    const long nK  = ((long)KHW * Cin) >> 5;          // K-steps of 32
    const long mstride = nK * 512;                    // elements per M-tile of Wf

    int p[2], oh[2], ow[2];
#pragma unroll
    for (int nt = 0; nt < 2; ++nt) {
        p[nt]  = n0 + nt * 16 + l15;
        oh[nt] = p[nt] / W;
        ow[nt] = p[nt] - oh[nt] * W;
    }

    f32x8 acc[4][2];
#pragma unroll
    for (int mt = 0; mt < 4; ++mt)
#pragma unroll
        for (int nt = 0; nt < 2; ++nt)
#pragma unroll
            for (int e = 0; e < 8; ++e) acc[mt][nt][e] = 0.0f;

    // Base of this block's A fragments: (m0/16)*mstride + lane*16
    const __bf16* abase0 = Wf + (long)(m0 >> 4) * mstride + (long)lane * 16;

    for (int t = 0; t < KHW; ++t) {
        const int kh = t / KW;
        const int kw = t - kh * KW;
        bool inb[2];
        const __bf16* xrow[2];
#pragma unroll
        for (int nt = 0; nt < 2; ++nt) {
            const int ih = oh[nt] + kh - padh;
            const int iw = ow[nt] + kw - padw;
            inb[nt]  = ((unsigned)ih < (unsigned)H) && ((unsigned)iw < (unsigned)W);
            xrow[nt] = Xp + (((long)b * H + ih) * W + iw) * Cin;
        }
        for (int cb = 0; cb < Cin; cb += 32) {
            // B fragments (32x16, K x N): lane holds col N=l15; element e -> K = hl*16+e
            bf16x16 bfrag[2];
#pragma unroll
            for (int nt = 0; nt < 2; ++nt) {
                if (inb[nt]) {
                    bfrag[nt] = *(const bf16x16*)(xrow[nt] + cb + hl * 16);
                } else {
#pragma unroll
                    for (int e = 0; e < 16; ++e) bfrag[nt][e] = (__bf16)0.0f;
                }
            }
            const long ks = ((long)t * Cin + cb) >> 5;
            const __bf16* aks = abase0 + ks * 512;
#pragma unroll
            for (int mt = 0; mt < 4; ++mt) {
                // A fragment: single contiguous 32B load (fragment-native layout)
                bf16x16 afrag = *(const bf16x16*)(aks + (long)mt * mstride);
                acc[mt][0] = __builtin_amdgcn_wmma_f32_16x16x32_bf16(
                    false, afrag, false, bfrag[0], (short)0, acc[mt][0], false, false);
                acc[mt][1] = __builtin_amdgcn_wmma_f32_16x16x32_bf16(
                    false, afrag, false, bfrag[1], (short)0, acc[mt][1], false, false);
            }
        }
    }

    // Epilogue + store (D layout: VGPR r -> M = hl*8 + r ; N = l15)
    const float rsq = 0.99999500003750f;   // 1/sqrt(1 + 1e-5)
#pragma unroll
    for (int mt = 0; mt < 4; ++mt) {
#pragma unroll
        for (int r = 0; r < 8; ++r) {
            const int m = m0 + mt * 16 + hl * 8 + r;
            float scale = gamma ? gamma[m] * rsq : 1.0f;
            float shift = beta ? beta[m] : 0.0f;
#pragma unroll
            for (int nt = 0; nt < 2; ++nt) {
                float v = acc[mt][nt][r] * scale + shift;
                if (relu) v = fmaxf(v, 0.0f);
                Y[((long)b * Cout + m) * HW + p[nt]] = v;
            }
        }
    }
}

// ---------------------------------------------------------------------------
// Pack f32 NCHW -> bf16 NHWC with channel offset (also implements concat).
// ---------------------------------------------------------------------------
__global__ void pack_nhwc_bf16(const float* __restrict__ src, __bf16* __restrict__ dst,
                               int Cs, int H, int W, int Cd, int coff, long total)
{
    long i = (long)blockIdx.x * blockDim.x + threadIdx.x;
    if (i >= total) return;
    int x = (int)(i % W); long t = i / W;
    int y = (int)(t % H); t /= H;
    int c = (int)(t % Cs);
    int b = (int)(t / Cs);
    dst[(((long)b * H + y) * W + x) * Cd + coff + c] = (__bf16)src[i];
}

// ---------------------------------------------------------------------------
// Pack weights f32 [Cout,Cin,KH,KW] -> WMMA A-fragment layout bf16:
//   dst[((co/16)*nK + ks)*32 + lane]*16 + e,  k = (kh*KW+kw)*Cin + ci,
//   ks = k/32, kk = k%32: g = 0, hl = kk>>4, sub = kk&15
//   lane = hl*16 + (co&15), e = g*8 + sub
// ---------------------------------------------------------------------------
__global__ void pack_w_frag(const float* __restrict__ src, __bf16* __restrict__ dst,
                            int Cin, int KH, int KW, long total)
{
    long i = (long)blockIdx.x * blockDim.x + threadIdx.x;
    if (i >= total) return;
    int kw = (int)(i % KW); long t = i / KW;
    int kh = (int)(t % KH); t /= KH;
    int ci = (int)(t % Cin);
    int co = (int)(t / Cin);
    const long k  = ((long)kh * KW + kw) * Cin + ci;
    const long ks = k >> 5;
    const int  kk = (int)(k & 31);
    const int  g  = 0;
    const int  hl = kk >> 4;
    const int  sub = kk & 15;
    const int  lane = hl * 16 + (co & 15);
    const int  e = g * 8 + sub;
    const long nK = ((long)KH * KW * Cin) >> 5;
    dst[(((long)(co >> 4) * nK + ks) * 32 + lane) * 16 + e] = (__bf16)src[i];
}

// ---------------------------------------------------------------------------
// 1x1 conv to a single output channel (pred_m1 / pred_m2 / out heads)
// ---------------------------------------------------------------------------
__global__ void conv_to1_kernel(const float* __restrict__ x, const float* __restrict__ w,
                                const float* __restrict__ bias, float* __restrict__ out,
                                int Cin, int HW, long total)
{
    long i = (long)blockIdx.x * blockDim.x + threadIdx.x;
    if (i >= total) return;
    long b = i / HW;
    int  p = (int)(i - b * HW);
    const float* xb = x + (long)b * Cin * HW + p;
    float s = bias[0];
    for (int c = 0; c < Cin; ++c) s += w[c] * xb[(long)c * HW];
    out[i] = s;
}

// ---------------------------------------------------------------------------
// sigmoid(mid1) -> 8x8 patch max -> mask -> stable argsort of (mask?0:1) -> k
// One block per batch, 64 threads (one per patch).
// ---------------------------------------------------------------------------
__global__ void mask_order_kernel(const float* __restrict__ mid1,
                                  int* __restrict__ kArr, int* __restrict__ order)
{
    __shared__ int msk[64];
    const int b = blockIdx.x;
    const int t = threadIdx.x;
    const int py = t >> 3, px = t & 7;
    const float* m = mid1 + (long)b * 64 * 64;
    float mx = -1e30f;
    for (int yy = 0; yy < 8; ++yy)
        for (int xx = 0; xx < 8; ++xx)
            mx = fmaxf(mx, m[(py * 8 + yy) * 64 + px * 8 + xx]);
    const float sig = 1.0f / (1.0f + __expf(-mx));
    msk[t] = (sig > 0.5f) ? 1 : 0;
    __syncthreads();
    if (t == 0) {
        int k = 0;
        for (int i = 0; i < 64; ++i) if (msk[i])  order[b * 64 + (k++)] = i;
        kArr[b] = k;
        int k2 = k;
        for (int i = 0; i < 64; ++i) if (!msk[i]) order[b * 64 + (k2++)] = i;
    }
}

// ---------------------------------------------------------------------------
// extra(o): gather TC=192 channels of bilinearly-resized (128x128 -> o x o,
// align-corners) raw_x patches selected by mask order. dst f32 [B,192,o,o].
// ---------------------------------------------------------------------------
__global__ void extra_kernel(const float* __restrict__ raw, const int* __restrict__ kArr,
                             const int* __restrict__ order, float* __restrict__ dst,
                             int o, long total)
{
    long i = (long)blockIdx.x * blockDim.x + threadIdx.x;
    if (i >= total) return;
    int x = (int)(i % o); long t = i / o;
    int y = (int)(t % o); t /= o;
    int j = (int)(t % 192);
    int b = (int)(t / 192);
    const int k = kArr[b];
    float v = 0.0f;
    if (k > 0) {
        const int denom = 3 * k;               // max(3k,1) == 3k when k>0
        const int s = j % denom;
        const int pidx = order[b * 64 + s / 3];
        const int ch = s - (s / 3) * 3;
        const int py = pidx >> 3, px = pidx & 7;
        const float sc = 127.0f / (float)(o - 1);
        const float ys = y * sc, xs = x * sc;
        int y0 = (int)ys; int y1 = imini(y0 + 1, 127); float wy = ys - y0;
        int x0 = (int)xs; int x1 = imini(x0 + 1, 127); float wx = xs - x0;
        const float* base = raw + ((long)b * 3 + ch) * 1024 * 1024
                                + (long)(py * 128) * 1024 + px * 128;
        float v00 = base[(long)y0 * 1024 + x0], v01 = base[(long)y0 * 1024 + x1];
        float v10 = base[(long)y1 * 1024 + x0], v11 = base[(long)y1 * 1024 + x1];
        float vy0 = v00 + (v10 - v00) * wy;
        float vy1 = v01 + (v11 - v01) * wy;
        v = vy0 + (vy1 - vy0) * wx;
    }
    dst[i] = v;
}

// ---------------------------------------------------------------------------
// Align-corners bilinear resize, f32 NCHW -> f32 NCHW.
// ---------------------------------------------------------------------------
__global__ void resize_kernel(const float* __restrict__ src, float* __restrict__ dst,
                              int C, int Hs, int Ws, int Hd, int Wd, long total)
{
    long i = (long)blockIdx.x * blockDim.x + threadIdx.x;
    if (i >= total) return;
    int x = (int)(i % Wd); long t = i / Wd;
    int y = (int)(t % Hd); t /= Hd;
    int c = (int)(t % C);
    int b = (int)(t / C);
    const float sy = (float)(Hs - 1) / (float)(Hd - 1);
    const float sx = (float)(Ws - 1) / (float)(Wd - 1);
    const float ys = y * sy, xs = x * sx;
    int y0 = (int)ys; int y1 = imini(y0 + 1, Hs - 1); float wy = ys - y0;
    int x0 = (int)xs; int x1 = imini(x0 + 1, Ws - 1); float wx = xs - x0;
    const float* s = src + (((long)b * C + c) * Hs) * Ws;
    float v00 = s[(long)y0 * Ws + x0], v01 = s[(long)y0 * Ws + x1];
    float v10 = s[(long)y1 * Ws + x0], v11 = s[(long)y1 * Ws + x1];
    float vy0 = v00 + (v10 - v00) * wy;
    float vy1 = v01 + (v11 - v01) * wy;
    dst[i] = vy0 + (vy1 - vy0) * wx;
}

// ---------------------------------------------------------------------------
// Host orchestration
// ---------------------------------------------------------------------------
extern "C" void kernel_launch(void* const* d_in, const int* in_sizes, int n_in,
                              void* d_out, int out_size, void* d_ws, size_t ws_size,
                              hipStream_t stream)
{
    (void)in_sizes; (void)n_in; (void)out_size; (void)ws_size;
    const float* coarse_m = (const float*)d_in[0];   // [4,128,64,64]
    const float* x2_2     = (const float*)d_in[1];   // [4,128,64,64]
    const float* x3_1     = (const float*)d_in[2];   // [4, 64,64,64]
    const float* raw_x    = (const float*)d_in[3];   // [4,  3,1024,1024]
    int idx = 4;
    auto NP_ = [&]() { return (const float*)d_in[idx++]; };
    const float* pred_m1_w = NP_(); const float* pred_m1_b = NP_();
    const float* dec1_c1_w = NP_(); const float* dec1_c1_g = NP_(); const float* dec1_c1_b = NP_();
    const float* dec1_c2_w = NP_(); const float* dec1_c2_g = NP_(); const float* dec1_c2_b = NP_();
    const float* dec2_c1_w = NP_(); const float* dec2_c1_g = NP_(); const float* dec2_c1_b = NP_();
    const float* dec2_c2_w = NP_(); const float* dec2_c2_g = NP_(); const float* dec2_c2_b = NP_();
    const float* cat1_w = NP_(); const float* cat1_g = NP_(); const float* cat1_b = NP_();
    const float* rs1_w  = NP_(); const float* rs1_b  = NP_();
    const float* up1_w  = NP_(); const float* up1_g  = NP_(); const float* up1_b = NP_();
    const float* pred_m2_w = NP_(); const float* pred_m2_b = NP_();
    const float* cat2_w = NP_(); const float* cat2_g = NP_(); const float* cat2_b = NP_();
    const float* rs2_w  = NP_(); const float* rs2_b  = NP_();
    const float* out_w  = NP_(); const float* out_b  = NP_();

    // Workspace arena
    char* base = (char*)d_ws;
    size_t off = 0;
    auto alloc = [&](size_t bytes) -> char* {
        char* p = base + off;
        off = (off + bytes + 255) & ~(size_t)255;
        return p;
    };
    float* R1 = (float*)alloc(4L * 512 * 128 * 128 * 4);   // big f32 stage (134 MB)
    float* R2 = (float*)alloc(4L * 128 * 128 * 128 * 4);   // mid f32 stage  (33.5 MB)
    float* R3 = (float*)alloc(4L * 192 * 128 * 128 * 4);   // extras/resizes (50.3 MB)
    __bf16* RP = (__bf16*)alloc(4L * 128 * 128 * 512 * 2); // packed NHWC bf16 (67 MB)
    __bf16* w_dec1c1 = (__bf16*)alloc(320L * 9 * 320 * 2);
    __bf16* w_dec1c2 = (__bf16*)alloc(320L * 9 * 320 * 2);
    __bf16* w_dec2c1 = (__bf16*)alloc(320L * 9 * 320 * 2);
    __bf16* w_dec2c2 = (__bf16*)alloc(320L * 9 * 320 * 2);
    __bf16* w_cat1   = (__bf16*)alloc(512L * 9 * 512 * 2);
    __bf16* w_up1    = (__bf16*)alloc(128L * 9 * 128 * 2);
    __bf16* w_cat2   = (__bf16*)alloc(512L * 25 * 512 * 2);
    __bf16* w_rs1    = (__bf16*)alloc(128L * 512 * 2);
    __bf16* w_rs2    = (__bf16*)alloc(128L * 512 * 2);
    int*   kArr      = (int*)alloc(4 * sizeof(int));
    int*   order     = (int*)alloc(256 * sizeof(int));
    float* out_small = (float*)alloc(4L * 128 * 128 * 4);
    float* x3r       = R3 + 4L * 128 * 128 * 128;          // tail of R3 (16.8 MB)

    float* outp = (float*)d_out;                    // [4,1,512,512]
    float* mid1 = outp + 4L * 512 * 512;            // [4,1, 64, 64]
    float* mid2 = mid1 + 4L * 64 * 64;              // [4,1,128,128]

    auto cdiv = [](long a, long b) { return (unsigned)((a + b - 1) / b); };
    auto packW = [&](const float* s, __bf16* d, int Cout, int Cin, int K) {
        long tot = (long)Cout * Cin * K * K;
        pack_w_frag<<<cdiv(tot, 256), 256, 0, stream>>>(s, d, Cin, K, K, tot);
    };
    auto pack = [&](const float* s, __bf16* d, int Cs, int H, int Cd, int coff) {
        long tot = 4L * Cs * H * H;
        pack_nhwc_bf16<<<cdiv(tot, 256), 256, 0, stream>>>(s, d, Cs, H, H, Cd, coff, tot);
    };
    auto conv = [&](const __bf16* X, const __bf16* Wt, float* Y, const float* g,
                    const float* bt, int relu, int Cin, int Cout, int H, int K) {
        dim3 grid((unsigned)(H * H / 128), (unsigned)(Cout / 64), 4);
        conv_wmma_kernel<<<grid, 128, 0, stream>>>(X, Wt, Y, g, bt, relu, Cin, Cout, H, H, K, K);
    };
    auto resize = [&](const float* s, float* d, int C, int Hs, int Hd) {
        long tot = 4L * C * Hd * Hd;
        resize_kernel<<<cdiv(tot, 256), 256, 0, stream>>>(s, d, C, Hs, Hs, Hd, Hd, tot);
    };

    // --- weight packing (deterministic, every call) ---
    packW(dec1_c1_w, w_dec1c1, 320, 320, 3);
    packW(dec1_c2_w, w_dec1c2, 320, 320, 3);
    packW(dec2_c1_w, w_dec2c1, 320, 320, 3);
    packW(dec2_c2_w, w_dec2c2, 320, 320, 3);
    packW(cat1_w,    w_cat1,   512, 512, 3);
    packW(up1_w,     w_up1,    128, 128, 3);
    packW(cat2_w,    w_cat2,   512, 512, 5);
    packW(rs1_w,     w_rs1,    128, 512, 1);
    packW(rs2_w,     w_rs2,    128, 512, 1);

    // --- mid1, mask/order, extra(64) ---
    conv_to1_kernel<<<cdiv(16384, 256), 256, 0, stream>>>(coarse_m, pred_m1_w, pred_m1_b,
                                                          mid1, 128, 4096, 16384);
    mask_order_kernel<<<4, 64, 0, stream>>>(mid1, kArr, order);
    {
        long tot = 4L * 192 * 64 * 64;
        extra_kernel<<<cdiv(tot, 256), 256, 0, stream>>>(raw_x, kArr, order, R3, 64, tot);
    }

    // --- dec1 block @64x64 ---
    pack(coarse_m, RP, 128, 64, 320, 0);
    pack(R3,       RP, 192, 64, 320, 128);
    conv(RP, w_dec1c1, R1, dec1_c1_g, dec1_c1_b, 1, 320, 320, 64, 3);
    pack(R1, RP, 320, 64, 320, 0);
    conv(RP, w_dec1c2, R1, dec1_c2_g, dec1_c2_b, 0, 320, 320, 64, 3);

    // --- cat1 (concat x2_2 | dec1 | x3_1) + rs1 ---
    pack(x2_2, RP, 128, 64, 512, 0);
    pack(R1,   RP, 320, 64, 512, 128);
    pack(x3_1, RP,  64, 64, 512, 448);
    conv(RP, w_cat1, R1, cat1_g, cat1_b, 0, 512, 512, 64, 3);
    pack(R1, RP, 512, 64, 512, 0);
    conv(RP, w_rs1, R2, nullptr, rs1_b, 0, 512, 128, 64, 1);

    // --- resize to 128, up1, mid2 ---
    resize(R2, R3, 128, 64, 128);
    pack(R3, RP, 128, 128, 128, 0);
    conv(RP, w_up1, R2, up1_g, up1_b, 0, 128, 128, 128, 3);
    conv_to1_kernel<<<cdiv(65536, 256), 256, 0, stream>>>(R2, pred_m2_w, pred_m2_b,
                                                          mid2, 128, 16384, 65536);

    // --- dec2 block @128x128 ---
    {
        long tot = 4L * 192 * 128 * 128;
        extra_kernel<<<cdiv(tot, 256), 256, 0, stream>>>(raw_x, kArr, order, R3, 128, tot);
    }
    pack(R2, RP, 128, 128, 320, 0);
    pack(R3, RP, 192, 128, 320, 128);
    conv(RP, w_dec2c1, R1, dec2_c1_g, dec2_c1_b, 1, 320, 320, 128, 3);
    pack(R1, RP, 320, 128, 320, 0);
    conv(RP, w_dec2c2, R1, dec2_c2_g, dec2_c2_b, 0, 320, 320, 128, 3);

    // --- cat2 (concat x2r | dec2 | x3r) 5x5 + rs2 + out head ---
    resize(x2_2, R3,  128, 64, 128);
    resize(x3_1, x3r,  64, 64, 128);
    pack(R3,  RP, 128, 128, 512, 0);
    pack(R1,  RP, 320, 128, 512, 128);
    pack(x3r, RP,  64, 128, 512, 448);
    conv(RP, w_cat2, R1, cat2_g, cat2_b, 0, 512, 512, 128, 5);
    pack(R1, RP, 512, 128, 512, 0);
    conv(RP, w_rs2, R2, nullptr, rs2_b, 0, 512, 128, 128, 1);
    conv_to1_kernel<<<cdiv(65536, 256), 256, 0, stream>>>(R2, out_w, out_b,
                                                          out_small, 128, 16384, 65536);
    resize(out_small, outp, 1, 128, 512);
}
